// ResidualAttentionBlock_EViT_86620900426084
// MI455X (gfx1250) — matrix operand order repaired and checked by reading, not compiled, not run on hardware
//
#include <hip/hip_runtime.h>
#include <hip/hip_bf16.h>

typedef _Float16 f16;
typedef __attribute__((ext_vector_type(16))) _Float16 v16h;
typedef __attribute__((ext_vector_type(8)))  float    v8f;

union AFrag { v16h v; f16 h[16]; unsigned u[8]; };
union CFrag { v8f  v; float f[8]; };

#define TOK 197          // N
#define BATCH 256        // B
#define CH 768           // C
#define HEADS 12
#define HEADD 64
#define LKEEP 98         // L = ceil(0.5*(N-1))
#define NEWTOK 100       // L + 2

#define KPAD 208         // 13 * 16 key tiles
#define KPAD32 224       // keys padded to K-step granularity (7 * 32)

// A-fragment K pattern within a 32-wide K step (contiguous: k = 16*(lane/16) + element)
__device__ __forceinline__ int a_kpat(int v) { return v * 2; }

// ---------------------------------------------------------------- TDM support
#if __has_builtin(__builtin_amdgcn_tensor_load_to_lds) && __has_builtin(__builtin_amdgcn_s_wait_tensorcnt)
#define USE_TDM 1
typedef __attribute__((ext_vector_type(4))) unsigned int u32x4;
typedef __attribute__((ext_vector_type(8))) int          i32x8;
typedef __attribute__((ext_vector_type(4))) int          i32x4;

// Async DMA of a 2D f16 tile (rows x rowlen, row stride in elements) global -> LDS.
// D# per CDNA5 ISA 8.3/8.4; groups 2/3 (and trailing group) zero: tile_dim2 = 0 => <=2D.
__device__ __forceinline__ void tdm_load_2d_f16(const f16* gsrc, unsigned lds_byte_addr,
                                                unsigned rows, unsigned rowlen,
                                                unsigned long long row_stride_elems) {
    unsigned long long ga = (unsigned long long)(const void*)gsrc;
    u32x4 g0;
    g0[0] = 1u;                                         // count=1, user descriptor
    g0[1] = lds_byte_addr;                              // lds_addr
    g0[2] = (unsigned)(ga & 0xFFFFFFFFu);               // global_addr[31:0]
    g0[3] = (unsigned)((ga >> 32) & 0x1FFFFFFu) | (2u << 30);   // global_addr[56:32] | type=2

    i32x8 g1;
    g1[0] = (int)(1u << 16);                            // data_size = 1 (2 bytes)
    g1[1] = (int)((rowlen & 0xFFFFu) << 16);            // tensor_dim0[15:0]
    g1[2] = (int)(((rowlen >> 16) & 0xFFFFu) | ((rows & 0xFFFFu) << 16));   // dim0 hi | dim1 lo
    g1[3] = (int)(((rows >> 16) & 0xFFFFu) | ((rowlen & 0xFFFFu) << 16));   // dim1 hi | tile_dim0
    g1[4] = (int)(rows & 0xFFFFu);                      // tile_dim1 | tile_dim2=0
    g1[5] = (int)(unsigned)(row_stride_elems & 0xFFFFFFFFull);  // dim0_stride[31:0]
    g1[6] = (int)(unsigned)((row_stride_elems >> 32) & 0xFFFFull); // dim0_stride[47:32]
    g1[7] = 0;

    i32x4 gz4 = {0, 0, 0, 0};
    i32x8 gz8 = {0, 0, 0, 0, 0, 0, 0, 0};
    // amdgpu-toolchain (clang-23) 6-arg form: (g0, g1, g2, g3, g4, cpol)
    __builtin_amdgcn_tensor_load_to_lds(g0, g1, gz4, gz4, gz8, 0);
}
#endif

// ---------------------------------------------------------------- utilities
__global__ void f32_to_f16_kernel(const float* __restrict__ src, f16* __restrict__ dst, long n) {
    long i = (long)blockIdx.x * blockDim.x + threadIdx.x;
    if (i < n) dst[i] = (f16)src[i];
}

__global__ void zero_f32_kernel(float* __restrict__ p, long n) {
    long i = (long)blockIdx.x * blockDim.x + threadIdx.x;
    if (i < n) p[i] = 0.f;
}

// LayerNorm over last dim (C=768) -> f16 output. One block per token.
__global__ void ln_to_f16_kernel(const float* __restrict__ x, const float* __restrict__ g,
                                 const float* __restrict__ bta, f16* __restrict__ out, int C) {
    long tok = blockIdx.x;
    const float* xp = x + tok * C;
    __shared__ float red[256];
    float s = 0.f, s2 = 0.f;
    for (int c = threadIdx.x; c < C; c += blockDim.x) { float v = xp[c]; s += v; s2 += v * v; }
    red[threadIdx.x] = s; __syncthreads();
    for (int st = 128; st > 0; st >>= 1) { if (threadIdx.x < st) red[threadIdx.x] += red[threadIdx.x + st]; __syncthreads(); }
    float mean = red[0] / C; __syncthreads();
    red[threadIdx.x] = s2; __syncthreads();
    for (int st = 128; st > 0; st >>= 1) { if (threadIdx.x < st) red[threadIdx.x] += red[threadIdx.x + st]; __syncthreads(); }
    float var = red[0] / C - mean * mean;
    float rstd = rsqrtf(var + 1e-5f);
    for (int c = threadIdx.x; c < C; c += blockDim.x)
        out[tok * C + c] = (f16)((xp[c] - mean) * rstd * g[c] + bta[c]);
}

// ---------------------------------------------------------------- WMMA GEMM
// out[m, n] = sum_k A[m,k] * W[n,k] + bias[n]   (A: MxK f16 row-major, W: NdxK f16 row-major)
// One wave computes a 32(M) x 64(N) block: 2 A-frags, 4 B-frags, 8 accumulators.
// mode 1: quickGELU -> f16     mode 2: -> f16     mode 3: + res -> f32
__global__ void wmma_gemm_kernel(const f16* __restrict__ A, const f16* __restrict__ W,
                                 const float* __restrict__ bias,
                                 float* __restrict__ outF, f16* __restrict__ outH,
                                 const float* __restrict__ res,
                                 int M, int Nd, int K, int mode) {
    int wave = threadIdx.x >> 5;
    int lane = threadIdx.x & 31;
    int tiles_n = Nd >> 6;
    long tile = (long)blockIdx.x * (blockDim.x >> 5) + wave;
    long total = (long)(M >> 5) * tiles_n;
    if (tile >= total) return;
    int tm = (int)(tile / tiles_n);
    int tn = (int)(tile % tiles_n);
    int Mbase = tm << 5, Nbase = tn << 6;

    int akoff = (lane >> 4) << 4;   // 0 or 16
    int bkoff = (lane >> 4) << 4;   // 0 or 16

    const f16* a0p = A + (long)(Mbase + (lane & 15)) * K;
    const f16* a1p = a0p + (long)16 * K;
    const f16* bp  = W + (long)(Nbase + (lane & 15)) * K;

    CFrag c[2][4];
#pragma unroll
    for (int h = 0; h < 2; ++h)
#pragma unroll
        for (int j = 0; j < 4; ++j)
#pragma unroll
            for (int i = 0; i < 8; ++i) c[h][j].f[i] = 0.f;

    for (int kb = 0; kb < K; kb += 32) {
        AFrag a0, a1;
#pragma unroll
        for (int v = 0; v < 8; ++v) {
            int ka = kb + akoff + a_kpat(v);
            a0.u[v] = *(const unsigned*)(a0p + ka);
            a1.u[v] = *(const unsigned*)(a1p + ka);
        }
#pragma unroll
        for (int j = 0; j < 4; ++j) {
            AFrag bf;
            const f16* wp = bp + (long)(j * 16) * K + kb + bkoff;
#pragma unroll
            for (int v = 0; v < 8; ++v) bf.u[v] = *(const unsigned*)(wp + v * 2);
            c[0][j].v = __builtin_amdgcn_wmma_f32_16x16x32_f16(false, a0.v, false, bf.v,
                                                               (short)0, c[0][j].v, false, false);
            c[1][j].v = __builtin_amdgcn_wmma_f32_16x16x32_f16(false, a1.v, false, bf.v,
                                                               (short)0, c[1][j].v, false, false);
        }
    }

    // C/D layout: lane&15 = N, M = vgpr + 8*(lane>>4)
#pragma unroll
    for (int h = 0; h < 2; ++h) {
        int mrow = Mbase + h * 16 + ((lane >> 4) << 3);
#pragma unroll
        for (int j = 0; j < 4; ++j) {
            int nIdx = Nbase + j * 16 + (lane & 15);
            float bb = bias[nIdx];
#pragma unroll
            for (int v = 0; v < 8; ++v) {
                long off = (long)(mrow + v) * Nd + nIdx;
                float val = c[h][j].f[v] + bb;
                if (mode == 1) { val = val / (1.f + __expf(-1.702f * val)); outH[off] = (f16)val; }
                else if (mode == 2) { outH[off] = (f16)val; }
                else { outF[off] = val + res[off]; }
            }
        }
    }
}

// ---------------------------------------------------------------- WMMA attention
// One block (256 thr = 8 waves) per (b, h). qkv: token-major m=(n*B+b), 2304 wide (q|k|v).
// LDS: K [208][64] f16, V^T [64][224] f16 (zero-padded),
//      region C = V row staging [208][64] f16 (TDM) aliased with P [16][224] f32 + red [16][16].
__global__ void attention_wmma_kernel(const f16* __restrict__ qkv, f16* __restrict__ o,
                                      float* __restrict__ cls) {
    const int bh = blockIdx.x;
    const int b = bh / HEADS, h = bh % HEADS;
    const int tid = threadIdx.x;
    const int wave = tid >> 5, lane = tid & 31;

    extern __shared__ char smem[];
    f16*   Ksh = (f16*)smem;                       // [KPAD][64]
    f16*   Vt  = Ksh + KPAD * HEADD;               // [64][KPAD32]
    char*  rc  = (char*)(Vt + HEADD * KPAD32);     // region C (aliased phases)
    float* Psh = (float*)rc;                       // [16][KPAD32]
    float* red = Psh + 16 * KPAD32;                // [16][16]

    const long strideT = (long)BATCH * 3 * CH;     // token stride inside qkv
    const f16* qbase = qkv + (long)b * 3 * CH + h * HEADD;
    const f16* kbase = qbase + CH;
    const f16* vbase = qbase + 2 * CH;

#ifdef USE_TDM
    f16* Vrows = (f16*)rc;                         // [KPAD][64] staging (dead after Vt built)
    if (tid < 32) {                                // wave 0 drives both DMAs
        tdm_load_2d_f16(kbase, (unsigned)(size_t)Ksh,   TOK, HEADD, (unsigned long long)strideT);
        tdm_load_2d_f16(vbase, (unsigned)(size_t)Vrows, TOK, HEADD, (unsigned long long)strideT);
        __builtin_amdgcn_s_wait_tensorcnt(0);
    }
    for (int i = TOK * HEADD + tid; i < KPAD * HEADD; i += 256) Ksh[i] = (f16)0.f;  // zero pad keys
    __syncthreads();                               // TDM data visible to all waves
    for (int i = tid; i < HEADD * KPAD32; i += 256) {
        int e = i / KPAD32, key = i % KPAD32;
        Vt[i] = (key < TOK) ? Vrows[key * HEADD + e] : (f16)0.f;
    }
    __syncthreads();                               // Vrows dead; region C becomes P/red
#else
    for (int i = tid; i < KPAD * HEADD; i += 256) {
        int key = i >> 6, e = i & 63;
        Ksh[i] = (key < TOK) ? kbase[(long)key * strideT + e] : (f16)0.f;
    }
    for (int i = tid; i < HEADD * KPAD32; i += 256) {
        int e = i / KPAD32, key = i % KPAD32;
        Vt[i] = (key < TOK) ? vbase[(long)key * strideT + e] : (f16)0.f;
    }
    __syncthreads();
#endif

    const int akoff = (lane >> 4) << 4;
    const int bkoff = (lane >> 4) << 4;
    const float scale = 0.125f;                    // 1/sqrt(64)
    const float invH = 1.f / (float)HEADS;

    for (int qt = 0; qt < 13; ++qt) {
        // Q fragments for this 16-query tile (2 K-steps over 64 features), kept in VGPRs
        int qrow = qt * 16 + (lane & 15); if (qrow > TOK - 1) qrow = TOK - 1;
        const f16* qp = qbase + (long)qrow * strideT;
        AFrag qa[2];
#pragma unroll
        for (int s = 0; s < 2; ++s)
#pragma unroll
            for (int v = 0; v < 8; ++v)
                qa[s].u[v] = *(const unsigned*)(qp + 32 * s + akoff + a_kpat(v));

        // ---- scores S = Q K^T * scale : 8 waves cover 13 key tiles
        for (int nt = wave; nt < 13; nt += 8) {
            CFrag c;
#pragma unroll
            for (int i = 0; i < 8; ++i) c.f[i] = 0.f;
#pragma unroll
            for (int s = 0; s < 2; ++s) {
                AFrag kf;
                const f16* kp = Ksh + (nt * 16 + (lane & 15)) * HEADD + 32 * s + bkoff;
#pragma unroll
                for (int v = 0; v < 8; ++v) kf.u[v] = *(const unsigned*)(kp + v * 2);
                c.v = __builtin_amdgcn_wmma_f32_16x16x32_f16(false, qa[s].v, false, kf.v,
                                                             (short)0, c.v, false, false);
            }
            int coln = nt * 16 + (lane & 15);
            int mr = (lane >> 4) << 3;
#pragma unroll
            for (int v = 0; v < 8; ++v) Psh[(mr + v) * KPAD32 + coln] = c.f[v] * scale;
        }
        __syncthreads();

        // ---- row softmax (16 threads per row); pad columns forced to exact zero
        {
            int row = tid >> 4, i = tid & 15;
            float mx = -3.4e38f;
            for (int col = i; col < TOK; col += 16) mx = fmaxf(mx, Psh[row * KPAD32 + col]);
            red[row * 16 + i] = mx;
            __syncthreads();
            mx = red[row * 16];
            for (int j = 1; j < 16; ++j) mx = fmaxf(mx, red[row * 16 + j]);
            float sum = 0.f;
            for (int col = i; col < KPAD32; col += 16) {
                float p = 0.f;
                if (col < TOK) { p = __expf(Psh[row * KPAD32 + col] - mx); sum += p; }
                Psh[row * KPAD32 + col] = p;
            }
            __syncthreads();                 // separates red reads (max) from red writes (sum)
            red[row * 16 + i] = sum;
            __syncthreads();
            sum = 0.f;
            for (int j = 0; j < 16; ++j) sum += red[row * 16 + j];
            float inv = 1.f / sum;
            for (int col = i; col < TOK; col += 16) Psh[row * KPAD32 + col] *= inv;
            if (qt == 0 && row == 0) {       // head-averaged CLS->patch attention
                for (int col = i; col < TOK; col += 16)
                    if (col >= 1) atomicAdd(&cls[(long)b * (TOK - 1) + col - 1],
                                            Psh[col] * invH);
            }
        }
        __syncthreads();

        // ---- O = P (16x224) x V^T : 4 feature tiles, waves 0..3, 7 K-steps
        if (wave < 4) {
            CFrag c;
#pragma unroll
            for (int i = 0; i < 8; ++i) c.f[i] = 0.f;
            int row = lane & 15;
            for (int ks = 0; ks < 7; ++ks) {
                AFrag pa, vf;
#pragma unroll
                for (int v = 0; v < 8; ++v) {
                    int ka = ks * 32 + akoff + a_kpat(v);
                    pa.h[2 * v]     = (f16)Psh[row * KPAD32 + ka];
                    pa.h[2 * v + 1] = (f16)Psh[row * KPAD32 + ka + 1];
                }
                const f16* vp = Vt + (wave * 16 + (lane & 15)) * KPAD32 + ks * 32 + bkoff;
#pragma unroll
                for (int v = 0; v < 8; ++v) vf.u[v] = *(const unsigned*)(vp + v * 2);
                c.v = __builtin_amdgcn_wmma_f32_16x16x32_f16(false, pa.v, false, vf.v,
                                                             (short)0, c.v, false, false);
            }
            int e = h * HEADD + wave * 16 + (lane & 15);
            int mr = (lane >> 4) << 3;
#pragma unroll
            for (int v = 0; v < 8; ++v) {
                int q = qt * 16 + mr + v;
                if (q < TOK) o[((long)q * BATCH + b) * CH + e] = (f16)c.f[v];
            }
        }
        __syncthreads();
    }
}

// ---------------------------------------------------------------- top-k + complement
__global__ void topk_kernel(const float* __restrict__ cls, int* __restrict__ idx,
                            int* __restrict__ comp_) {
    int b = blockIdx.x;
    const float* c = cls + (long)b * (TOK - 1);
    bool taken[TOK - 1];
    float v[TOK - 1];
    for (int i = 0; i < TOK - 1; ++i) { taken[i] = false; v[i] = c[i]; }
    int* ib = idx + b * LKEEP;
    int* cb = comp_ + b * LKEEP;
    for (int i = 0; i < LKEEP; ++i) {        // stable descending (ties -> lower index)
        int best = 0; float bv = -3.4e38f;
        for (int j = 0; j < TOK - 1; ++j)
            if (!taken[j] && v[j] > bv) { bv = v[j]; best = j; }
        ib[i] = best; taken[best] = true;
    }
    int pos = 0;
    for (int j = 0; j < TOK - 1; ++j) if (!taken[j]) cb[pos++] = j;
}

// ---------------------------------------------------------------- EViT token rebuild
__global__ void build_xnew_kernel(const float* __restrict__ xres, const float* __restrict__ cls,
                                  const int* __restrict__ idx, const int* __restrict__ comp_,
                                  float* __restrict__ xnew) {
    int b = blockIdx.x, row = blockIdx.y;
    if (row == 0) {
        for (int c = threadIdx.x; c < CH; c += blockDim.x)
            xnew[((long)0 * BATCH + b) * CH + c] = xres[((long)0 * BATCH + b) * CH + c];
    } else if (row <= LKEEP) {
        int tk = idx[b * LKEEP + row - 1];
        for (int c = threadIdx.x; c < CH; c += blockDim.x)
            xnew[((long)row * BATCH + b) * CH + c] = xres[((long)(tk + 1) * BATCH + b) * CH + c];
    } else {
        __shared__ int   ci[LKEEP];
        __shared__ float cw[LKEEP];
        if (threadIdx.x < LKEEP) {
            int j = comp_[b * LKEEP + threadIdx.x];
            ci[threadIdx.x] = j;
            cw[threadIdx.x] = cls[(long)b * (TOK - 1) + j];
        }
        __syncthreads();
        for (int c = threadIdx.x; c < CH; c += blockDim.x) {
            float acc = 0.f;
            for (int j = 0; j < LKEEP; ++j)
                acc += cw[j] * xres[((long)(ci[j] + 1) * BATCH + b) * CH + c];
            xnew[((long)(LKEEP + 1) * BATCH + b) * CH + c] = acc;
        }
    }
}

// ---------------------------------------------------------------- host launcher
extern "C" void kernel_launch(void* const* d_in, const int* in_sizes, int n_in,
                              void* d_out, int out_size, void* d_ws, size_t ws_size,
                              hipStream_t stream) {
    (void)in_sizes; (void)n_in; (void)out_size; (void)ws_size;
    const float* x         = (const float*)d_in[0];
    const float* ln1_g     = (const float*)d_in[1];
    const float* ln1_b     = (const float*)d_in[2];
    const float* in_proj_w = (const float*)d_in[3];
    const float* in_proj_b = (const float*)d_in[4];
    const float* out_proj_w= (const float*)d_in[5];
    const float* out_proj_b= (const float*)d_in[6];
    const float* ln2_g     = (const float*)d_in[7];
    const float* ln2_b     = (const float*)d_in[8];
    const float* fc_w      = (const float*)d_in[9];
    const float* fc_b      = (const float*)d_in[10];
    const float* proj_w    = (const float*)d_in[11];
    const float* proj_b    = (const float*)d_in[12];

    const long MTOK = (long)TOK * BATCH;            // 50432
    const long MNEW = (long)NEWTOK * BATCH;         // 25600

    // ---- workspace layout (with reuse)
    char* base = (char*)d_ws;
    size_t off = 0;
    auto take = [&](size_t bytes) -> char* {
        off = (off + 255) & ~(size_t)255;
        char* p = base + off; off += bytes; return p;
    };
    f16*  xln   = (f16*)take(MTOK * CH * sizeof(f16));          // reused as attention out (same size)
    f16*  w16   = (f16*)take((size_t)(3*CH + CH + 4*CH + 4*CH) * CH * sizeof(f16));
    f16*  qkvh  = (f16*)take(MTOK * 3 * CH * sizeof(f16));      // reused as gelu(fc) f16 (smaller)
    float* cls  = (float*)take((size_t)BATCH * (TOK - 1) * sizeof(float));
    float* xres = (float*)take(MTOK * CH * sizeof(float));      // reused as LN2 f16 (smaller)
    int*  idx   = (int*)take((size_t)BATCH * LKEEP * sizeof(int));
    int*  comp_ = (int*)take((size_t)BATCH * LKEEP * sizeof(int));
    float* xnew = (float*)take(MNEW * CH * sizeof(float));

    f16* w_in  = w16;
    f16* w_out = w_in  + (size_t)3 * CH * CH;
    f16* w_fc  = w_out + (size_t)CH * CH;
    f16* w_pr  = w_fc  + (size_t)4 * CH * CH;
    f16* oh    = xln;                 // alias: attention output (f16, MTOK x CH)
    f16* fch   = qkvh;                // alias: gelu(fc) (f16, MNEW x 4CH)
    f16* xnln  = (f16*)xres;          // alias: LN2 output (f16, MNEW x CH)

    auto cvt = [&](const float* s, f16* d, long n) {
        f32_to_f16_kernel<<<(unsigned)((n + 255) / 256), 256, 0, stream>>>(s, d, n);
    };
    auto gemm = [&](const f16* A, const f16* W, const float* bias, float* oF, f16* oH,
                    const float* res, long M, int Nd, int K, int mode) {
        long tiles = (M >> 5) * (Nd >> 6);
        wmma_gemm_kernel<<<(unsigned)((tiles + 7) / 8), 256, 0, stream>>>(
            A, W, bias, oF, oH, res, (int)M, Nd, K, mode);
    };

    // 1. weights -> f16
    cvt(in_proj_w,  w_in,  (long)3 * CH * CH);
    cvt(out_proj_w, w_out, (long)CH * CH);
    cvt(fc_w,       w_fc,  (long)4 * CH * CH);
    cvt(proj_w,     w_pr,  (long)CH * 4 * CH);

    // 2. LN1 -> f16
    ln_to_f16_kernel<<<(unsigned)MTOK, 256, 0, stream>>>(x, ln1_g, ln1_b, xln, CH);

    // 3. zero cls-attn accumulator
    zero_f32_kernel<<<(unsigned)((BATCH * (TOK - 1) + 255) / 256), 256, 0, stream>>>(
        cls, (long)BATCH * (TOK - 1));

    // 4. QKV GEMM -> f16
    gemm(xln, w_in, in_proj_b, nullptr, qkvh, nullptr, MTOK, 3 * CH, CH, 2);

    // 5. WMMA attention (TDM-staged K/V; also accumulates head-averaged CLS attention)
    {
        // K tile + V^T tile + region C (max of V row staging, P + reductions)
        size_t smem = (size_t)(KPAD * HEADD + HEADD * KPAD32) * sizeof(f16)
                    + (size_t)(KPAD * HEADD) * sizeof(f16);    // 81920 B
        attention_wmma_kernel<<<BATCH * HEADS, 256, smem, stream>>>(qkvh, oh, cls);
    }

    // 6. out-proj GEMM + residual -> xres (f32)
    gemm(oh, w_out, out_proj_b, xres, nullptr, x, MTOK, CH, CH, 3);

    // 7. top-k + complement per batch
    topk_kernel<<<BATCH, 1, 0, stream>>>(cls, idx, comp_);

    // 8. rebuild token set (cls | top-k | weighted merge)
    build_xnew_kernel<<<dim3(BATCH, NEWTOK), 256, 0, stream>>>(xres, cls, idx, comp_, xnew);

    // 9. LN2 -> f16
    ln_to_f16_kernel<<<(unsigned)MNEW, 256, 0, stream>>>(xnew, ln2_g, ln2_b, xnln, CH);

    // 10. fc GEMM + quickGELU -> f16
    gemm(xnln, w_fc, fc_b, nullptr, fch, nullptr, MNEW, 4 * CH, CH, 1);

    // 11. proj GEMM + residual -> d_out (f32)
    gemm(fch, w_pr, proj_b, (float*)d_out, nullptr, xnew, MNEW, CH, 4 * CH, 3);
}
